// AttentionRNNCell_30374008717342
// MI455X (gfx1250) — compile-verified, matrix operand order checked by reading, not run on hardware
//
#include <hip/hip_runtime.h>

#define B_   128
#define TE_  2048
#define U_   256
#define IN_  256
#define MT_  4          // 16-row M-tiles per workgroup (64 timesteps)

typedef __attribute__((ext_vector_type(16))) __bf16 v16bf;
typedef __attribute__((ext_vector_type(8)))  float  v8f;

__device__ __forceinline__ unsigned short f32_to_bf16_bits(float f) {
  unsigned int u = __float_as_uint(f);
  u += 0x7FFFu + ((u >> 16) & 1u);          // round-to-nearest-even
  return (unsigned short)(u >> 16);
}

// ---------------------------------------------------------------------------
// Prep: U_a (256x256 f32, row-major [e][u]) -> bf16 transposed [u][e] so each
// WMMA B-fragment lane reads two contiguous 8-element (16B) runs.
// ---------------------------------------------------------------------------
__global__ void k_prep_uaT(const float* __restrict__ Ua,
                           unsigned short* __restrict__ UaT) {
  int e = blockIdx.x;                 // 0..255
  int u = threadIdx.x;                // 0..255
  UaT[u * U_ + e] = f32_to_bf16_bits(Ua[e * U_ + u]);
}

// wxstm = h_tm @ W_a   ([128,256] x [256,256], tiny)
__global__ void k_wxstm(const float* __restrict__ h,
                        const float* __restrict__ Wa,
                        float* __restrict__ wx) {
  int b = blockIdx.x, u = threadIdx.x;
  __shared__ float sh[U_];
  sh[u] = h[b * U_ + u];
  __syncthreads();
  float a = 0.f;
  #pragma unroll 4
  for (int k = 0; k < U_; ++k) a = fmaf(sh[k], Wa[k * U_ + u], a);
  wx[b * U_ + u] = a;
}

// ---------------------------------------------------------------------------
// Scores: et[b,t] = sum_u V_a[u]*tanh(wxstm[b,u] + b_a[u] + (x_seq[b,t,:]@U_a)[u])
// One workgroup = (b, 64-timestep) block: 4 M-tiles x 16 N-tiles.
// 8 waves x 2 N-tiles, fused in the K-loop: each A fragment read once from LDS
// feeds two WMMAs (two independent accumulator chains) -> 64 v_wmma per wave.
// ---------------------------------------------------------------------------
__global__ __launch_bounds__(256) void k_scores(
    const float* __restrict__ x_seq, const unsigned short* __restrict__ UaT,
    const float* __restrict__ wxstm, const float* __restrict__ b_a,
    const float* __restrict__ V_a, float* __restrict__ et) {
  __shared__ __align__(16) unsigned short sA[16 * MT_ * U_];  // 64 x 256 bf16 = 32KB
  __shared__ float s_e[16 * MT_];

  int b  = blockIdx.y;
  int t0 = blockIdx.x * (16 * MT_);
  int tid = threadIdx.x;

  // Stage A block (64 timesteps x 256 K) into LDS as bf16 (coalesced).
  const float* xrow = x_seq + ((size_t)b * TE_ + t0) * U_;
  for (int i = tid; i < 16 * MT_ * U_; i += 256) sA[i] = f32_to_bf16_bits(xrow[i]);
  if (tid < 16 * MT_) s_e[tid] = 0.f;
  __syncthreads();

  int wave = tid >> 5;
  int lane = tid & 31;
  int sel  = lane >> 4;      // 0: K runs {0..7},{16..23}; 1: {8..15},{24..31}
  int row  = lane & 15;      // A row (M) / B column (N)

  v8f acc[2][MT_];
  #pragma unroll
  for (int j = 0; j < 2; ++j)
    #pragma unroll
    for (int m = 0; m < MT_; ++m)
      acc[j][m] = (v8f){0.f, 0.f, 0.f, 0.f, 0.f, 0.f, 0.f, 0.f};

  const unsigned short* bcol0 = UaT + (size_t)(2 * wave * 16 + row) * U_;
  const unsigned short* bcol1 = bcol0 + 16 * U_;

  #pragma unroll
  for (int kk = 0; kk < U_; kk += 32) {
    union { uint4 q[2]; v16bf v; } fb0, fb1;
    fb0.q[0] = *(const uint4*)(bcol0 + kk + sel * 8);
    fb0.q[1] = *(const uint4*)(bcol0 + kk + 16 + sel * 8);
    fb1.q[0] = *(const uint4*)(bcol1 + kk + sel * 8);
    fb1.q[1] = *(const uint4*)(bcol1 + kk + 16 + sel * 8);
    #pragma unroll
    for (int m = 0; m < MT_; ++m) {
      union { uint4 q[2]; v16bf v; } fa;
      const unsigned short* arow = &sA[(m * 16 + row) * U_ + kk];
      fa.q[0] = *(const uint4*)(arow + sel * 8);
      fa.q[1] = *(const uint4*)(arow + 16 + sel * 8);
      acc[0][m] = __builtin_amdgcn_wmma_f32_16x16x32_bf16(
          false, fa.v, false, fb0.v, (short)0, acc[0][m], false, false);
      acc[1][m] = __builtin_amdgcn_wmma_f32_16x16x32_bf16(
          false, fa.v, false, fb1.v, (short)0, acc[1][m], false, false);
    }
  }

  // Epilogue: C layout -> (M = i + 8*sel, N = row). Fuse tanh + V_a dot.
  #pragma unroll
  for (int j = 0; j < 2; ++j) {
    int ug = (2 * wave + j) * 16 + row;
    float wx = wxstm[b * U_ + ug] + b_a[ug];
    float va = V_a[ug];
    #pragma unroll
    for (int m = 0; m < MT_; ++m) {
      #pragma unroll
      for (int i = 0; i < 8; ++i) {
        atomicAdd(&s_e[m * 16 + i + 8 * sel], va * tanhf(acc[j][m][i] + wx));
      }
    }
  }
  __syncthreads();
  if (tid < 16 * MT_) et[(size_t)b * TE_ + t0 + tid] = s_e[tid];
}

// ---------------------------------------------------------------------------
// Softmax over TE (unnormalized exp / sum, like reference). Also zero context.
// ---------------------------------------------------------------------------
__global__ void k_softmax(float* __restrict__ at, float* __restrict__ ctx) {
  int b = blockIdx.x, tid = threadIdx.x;
  float v[8];
  float s = 0.f;
  #pragma unroll
  for (int j = 0; j < 8; ++j) {
    v[j] = expf(at[(size_t)b * TE_ + j * 256 + tid]);
    s += v[j];
  }
  __shared__ float red[256];
  red[tid] = s;
  __syncthreads();
  for (int off = 128; off > 0; off >>= 1) {
    if (tid < off) red[tid] += red[tid + off];
    __syncthreads();
  }
  float inv = 1.f / red[0];
  #pragma unroll
  for (int j = 0; j < 8; ++j)
    at[(size_t)b * TE_ + j * 256 + tid] = v[j] * inv;
  ctx[b * U_ + tid] = 0.f;
}

// context[b,u] = sum_t at[b,t] * x_seq[b,t,u]  (t-segmented, global f32 atomics)
__global__ void k_context(const float* __restrict__ x_seq,
                          const float* __restrict__ at,
                          float* __restrict__ ctx) {
  int seg = blockIdx.x, b = blockIdx.y, u = threadIdx.x;
  const float* xp = x_seq + ((size_t)b * TE_ + seg * 128) * U_ + u;
  const float* ap = at + (size_t)b * TE_ + seg * 128;
  float acc = 0.f;
  for (int t = 0; t < 128; ++t) {
    __builtin_prefetch(xp + 8 * U_, 0, 1);
    acc = fmaf(ap[t], xp[0], acc);
    xp += U_;
  }
  atomicAdd(&ctx[b * U_ + u], acc);
}

// ---------------------------------------------------------------------------
// GRU gates: zt, rt (needs full rows, so split from the candidate kernel).
// ---------------------------------------------------------------------------
__global__ void k_gates(const float* __restrict__ h, const float* __restrict__ in,
                        const float* __restrict__ ctx,
                        const float* __restrict__ Wz, const float* __restrict__ Cz,
                        const float* __restrict__ bz,
                        const float* __restrict__ Wr, const float* __restrict__ Cr,
                        const float* __restrict__ br,
                        float* __restrict__ ztw, float* __restrict__ rhw) {
  int b = blockIdx.x, u = threadIdx.x;
  __shared__ float sh[U_], si[IN_], sc[U_];
  sh[u] = h[b * U_ + u];
  si[u] = in[b * IN_ + u];
  sc[u] = ctx[b * U_ + u];
  __syncthreads();
  float az = bz[u], ar = br[u];
  for (int k = 0; k < U_; ++k) {
    float hv = sh[k];
    az = fmaf(hv, Wz[k * U_ + u], az);
    ar = fmaf(hv, Wr[k * U_ + u], ar);
  }
  for (int k = 0; k < IN_; ++k) {
    float iv = si[k];
    az = fmaf(iv, Cz[k * U_ + u], az);
    ar = fmaf(iv, Cr[k * U_ + u], ar);
  }
  for (int k = 0; k < U_; ++k) {
    float cv = sc[k];
    az = fmaf(cv, Cz[(IN_ + k) * U_ + u], az);
    ar = fmaf(cv, Cr[(IN_ + k) * U_ + u], ar);
  }
  float zt = 1.f / (1.f + expf(-az));
  float rt = 1.f / (1.f + expf(-ar));
  ztw[b * U_ + u] = zt;
  rhw[b * U_ + u] = rt * sh[u];
}

// t_ht = tanh((rt*h)@U_p + ci@C_p + b_p);  ht = (1-zt)*h + zt*t_ht
__global__ void k_out(const float* __restrict__ h, const float* __restrict__ in,
                      const float* __restrict__ ctx,
                      const float* __restrict__ Up, const float* __restrict__ Cp,
                      const float* __restrict__ bp,
                      const float* __restrict__ ztw, const float* __restrict__ rhw,
                      float* __restrict__ out) {
  int b = blockIdx.x, u = threadIdx.x;
  __shared__ float sr[U_], si[IN_], sc[U_];
  sr[u] = rhw[b * U_ + u];
  si[u] = in[b * IN_ + u];
  sc[u] = ctx[b * U_ + u];
  __syncthreads();
  float a = bp[u];
  for (int k = 0; k < U_; ++k)  a = fmaf(sr[k], Up[k * U_ + u], a);
  for (int k = 0; k < IN_; ++k) a = fmaf(si[k], Cp[k * U_ + u], a);
  for (int k = 0; k < U_; ++k)  a = fmaf(sc[k], Cp[(IN_ + k) * U_ + u], a);
  float th = tanhf(a);
  float zt = ztw[b * U_ + u];
  out[b * U_ + u] = (1.f - zt) * h[b * U_ + u] + zt * th;
}

// ---------------------------------------------------------------------------
extern "C" void kernel_launch(void* const* d_in, const int* in_sizes, int n_in,
                              void* d_out, int out_size, void* d_ws, size_t ws_size,
                              hipStream_t stream) {
  const float* inputs = (const float*)d_in[0];
  const float* h_tm   = (const float*)d_in[1];
  const float* x_seq  = (const float*)d_in[2];
  const float* V_a    = (const float*)d_in[3];
  const float* W_a    = (const float*)d_in[4];
  const float* U_a    = (const float*)d_in[5];
  const float* b_a    = (const float*)d_in[6];
  const float* C_z    = (const float*)d_in[7];
  const float* W_z    = (const float*)d_in[8];
  const float* b_z    = (const float*)d_in[9];
  const float* C_r    = (const float*)d_in[10];
  const float* W_r    = (const float*)d_in[11];
  const float* b_r    = (const float*)d_in[12];
  const float* C_p    = (const float*)d_in[13];
  const float* U_p    = (const float*)d_in[14];
  const float* b_p    = (const float*)d_in[15];
  float* out = (float*)d_out;

  // Workspace layout (floats): wxstm | et/at | context | zt | rh | UaT(bf16)
  float* wsf   = (float*)d_ws;
  float* wxstm = wsf;                              // 32768
  float* et    = wsf + 32768;                      // 262144 (reused as at)
  float* ctx   = et + (size_t)B_ * TE_;            // 32768
  float* ztw   = ctx + B_ * U_;                    // 32768
  float* rhw   = ztw + B_ * U_;                    // 32768
  unsigned short* UaT = (unsigned short*)(rhw + B_ * U_);  // 65536 bf16

  dim3 blk(256);
  k_prep_uaT<<<dim3(U_), blk, 0, stream>>>(U_a, UaT);
  k_wxstm  <<<dim3(B_), blk, 0, stream>>>(h_tm, W_a, wxstm);
  k_scores <<<dim3(TE_ / (16 * MT_), B_), blk, 0, stream>>>(x_seq, UaT, wxstm, b_a,
                                                            V_a, et);
  k_softmax<<<dim3(B_), blk, 0, stream>>>(et, ctx);
  k_context<<<dim3(TE_ / 128, B_), blk, 0, stream>>>(x_seq, et, ctx);
  k_gates  <<<dim3(B_), blk, 0, stream>>>(h_tm, inputs, ctx, W_z, C_z, b_z,
                                          W_r, C_r, b_r, ztw, rhw);
  k_out    <<<dim3(B_), blk, 0, stream>>>(h_tm, inputs, ctx, U_p, C_p, b_p,
                                          ztw, rhw, out);
}
